// TextGRURegressor_74062416052917
// MI455X (gfx1250) — compile-verified, matrix operand order checked by reading, not run on hardware
//
#include <hip/hip_runtime.h>
#include <hip/hip_bf16.h>
#include <math.h>

// ---------------------------------------------------------------------------
// TextGRURegressor on MI455X (gfx1250, wave32, WMMA bf16)
//   B=128, T=512, H=192 (3H=576), E=128
// Pipeline:
//   embed -> [bf16 weights, WMMA GEMM gx, persistent WMMA GRU scan] x3 -> head
// Scan exploits per-batch-row independence of the recurrence:
//   grid = 8 batch-groups x 2 directions, each WG persistent with w_hh held
//   as register-resident WMMA B-fragments and a ping-pong LDS h mirror.
// ---------------------------------------------------------------------------

#define Bb   128
#define Tt   512
#define Hh   192
#define G3H  576
#define Ee   128

typedef __attribute__((ext_vector_type(16))) __bf16 v16bf;
typedef __attribute__((ext_vector_type(8)))  __bf16 v8bf;
typedef __attribute__((ext_vector_type(8)))  float  v8f;

// Load a 16x32 bf16 WMMA A/B fragment for this lane from a contiguous row.
// Per ISA 7.12.2: lane half (L>>4) selects k-chunks [half*8, half*8+7] and
// [16+half*8, 16+half*8+7]  (two 16-byte loads).
__device__ __forceinline__ v16bf load_frag(const __bf16* __restrict__ p, int half) {
    v8bf lo = *reinterpret_cast<const v8bf*>(p + half * 8);
    v8bf hi = *reinterpret_cast<const v8bf*>(p + 16 + half * 8);
    v16bf r;
#pragma unroll
    for (int i = 0; i < 8; ++i) { r[i] = lo[i]; r[8 + i] = hi[i]; }
    return r;
}

__device__ __forceinline__ v8f wmma_bf16(v16bf a, v16bf b, v8f c) {
    // (neg_a, A, neg_b, B, c_mod, C, reuse_a, reuse_b)
    return __builtin_amdgcn_wmma_f32_16x16x32_bf16(false, a, false, b, (short)0, c,
                                                   false, false);
}

__device__ __forceinline__ float sigmoidf_(float x) { return 1.0f / (1.0f + expf(-x)); }

// ---------------------------------------------------------------------------
// fp32 -> bf16 weight conversion
// ---------------------------------------------------------------------------
__global__ void to_bf16_kernel(const float* __restrict__ in, __bf16* __restrict__ out,
                               int nel) {
    int i = blockIdx.x * 256 + threadIdx.x;
    if (i < nel) out[i] = (__bf16)in[i];
}

// ---------------------------------------------------------------------------
// Embedding gather -> bf16 activations in (T, B, E) layout
// ---------------------------------------------------------------------------
__global__ void embed_kernel(const int* __restrict__ x, const float* __restrict__ emb,
                             __bf16* __restrict__ act) {
    size_t tid = (size_t)blockIdx.x * 256 + threadIdx.x;   // t*B*E + b*E + e
    if (tid >= (size_t)Tt * Bb * Ee) return;
    int e       = (int)(tid & (Ee - 1));
    size_t tb   = tid >> 7;               // /E
    int b       = (int)(tb & (Bb - 1));
    int t       = (int)(tb >> 7);         // /B
    int idx     = x[(size_t)b * Tt + t];
    act[tid]    = (__bf16)emb[(size_t)idx * Ee + e];
}

// ---------------------------------------------------------------------------
// Input-projection GEMM (bf16 WMMA, f32 accum):
//   C[dir][m][n] = sum_k A[m][k] * W[dir][n][k] + bias[dir][n]
// A: [M][K] bf16 (row = t*B + b),  W: [2][N][K] bf16,  C: [2][M][N] f32
// One wave -> 16 (M) x 64 (N) tile: A frag reused across 4 B frags.
// ---------------------------------------------------------------------------
__global__ __launch_bounds__(256) void gemm_proj_kernel(
    const __bf16* __restrict__ A, const __bf16* __restrict__ Wt,
    const float* __restrict__ bias, float* __restrict__ C,
    int M, int N, int K) {
    const int dir = blockIdx.y;
    const __bf16* W  = Wt + (size_t)dir * N * K;
    const float*  bs = bias + (size_t)dir * N;
    float*        Cd = C + (size_t)dir * M * N;

    const int gw      = blockIdx.x * 8 + (threadIdx.x >> 5);
    const int ngroups = N / 64;
    const int mt      = gw / ngroups;
    const int ng      = gw - mt * ngroups;
    if (mt * 16 >= M) return;

    const int lane = threadIdx.x & 31;
    const int n    = lane & 15;
    const int half = lane >> 4;

    const __bf16* arow = A + (size_t)(mt * 16 + n) * K;

    v8f acc[4];
#pragma unroll
    for (int j = 0; j < 4; ++j)
#pragma unroll
        for (int v = 0; v < 8; ++v) acc[j][v] = 0.0f;

    for (int k0 = 0; k0 < K; k0 += 32) {
        v16bf a = load_frag(arow + k0, half);
#pragma unroll
        for (int j = 0; j < 4; ++j) {
            const __bf16* wrow = W + (size_t)(ng * 64 + j * 16 + n) * K;
            v16bf b = load_frag(wrow + k0, half);
            acc[j] = wmma_bf16(a, b, acc[j]);
        }
    }
#pragma unroll
    for (int j = 0; j < 4; ++j) {
        const int col = ng * 64 + j * 16 + n;
        const float bv = bs[col];
#pragma unroll
        for (int v = 0; v < 8; ++v) {
            const int row = mt * 16 + v + 8 * half;
            Cd[(size_t)row * N + col] = acc[j][v] + bv;
        }
    }
}

// ---------------------------------------------------------------------------
// Persistent bidirectional GRU scan, batch-split.
// grid = (8 batch-groups, 2 directions); 384 threads = 12 waves per WG.
// Wave w owns hidden slice [w*16, w*16+16) for this WG's 16 batch rows.
//  - w_hh kept as t-invariant WMMA B-fragments in registers (3 gates x 6 ksteps)
//  - h: f32 in registers (C layout) + ping-pong bf16 mirror in LDS (2 x 6 KB)
//    feeding WMMA A fragments; exactly ONE s_barrier per timestep
//  - per step per wave: 18 WMMAs (3 independent 6-deep chains) + fused gates
// gx:  [2][T][B][576] f32     out: [T][B][384] bf16 (dir writes cols dir*H..)
// ---------------------------------------------------------------------------
__global__ __launch_bounds__(384) void gru_scan_kernel(
    const float* __restrict__ gx_all, const __bf16* __restrict__ whh_all,
    const float* __restrict__ bhh_all, __bf16* __restrict__ out) {
    const int bg  = blockIdx.x;          // batch rows [bg*16, bg*16+16)
    const int dir = blockIdx.y;
    const float*  gx  = gx_all + (size_t)dir * Tt * Bb * G3H;
    const __bf16* wh  = whh_all + (size_t)dir * G3H * Hh;
    const float*  bhh = bhh_all + (size_t)dir * G3H;

    __shared__ __bf16 hsh[2][16 * Hh];   // ping-pong hidden mirror (2 x 6 KB)

    const int tid  = threadIdx.x;
    const int wid  = tid >> 5;           // 0..11
    const int lane = tid & 31;
    const int n    = lane & 15;
    const int half = lane >> 4;
    const int j0   = wid * 16;           // hidden slice owned by this wave
    const int b0   = bg * 16;            // first batch row of this WG

    for (int i = tid; i < 16 * Hh; i += 384) hsh[0][i] = (__bf16)0.0f;

    // t-invariant w_hh B-fragments + recurrent biases
    v16bf bfrag[3][6];
    float bb[3];
#pragma unroll
    for (int g = 0; g < 3; ++g) {
        const int col = g * Hh + j0 + n;
        const __bf16* wrow = wh + (size_t)col * Hh;
#pragma unroll
        for (int ks = 0; ks < 6; ++ks) bfrag[g][ks] = load_frag(wrow + ks * 32, half);
        bb[g] = bhh[col];
    }

    float hreg[8];                       // f32 hidden state, C layout
#pragma unroll
    for (int v = 0; v < 8; ++v) hreg[v] = 0.0f;

    __syncthreads();

    for (int s = 0; s < Tt; ++s) {
        const int t = dir ? (Tt - 1 - s) : s;
        const __bf16* hcur = hsh[s & 1];
        __bf16*       hnxt = hsh[(s + 1) & 1];
        const float*  gxt  = gx + ((size_t)t * Bb + b0) * G3H;

        // Prefetch next step's gx slice for this WG's rows
        // (16 rows * 576 * 4 B = 36,864 B = 384 threads * 96 B)
        if (s + 1 < Tt) {
            const int tn = dir ? (t - 1) : (t + 1);
            const char* nxt = (const char*)(gx + ((size_t)tn * Bb + b0) * G3H);
            __builtin_prefetch(nxt + (size_t)tid * 96, 0, 0);
            __builtin_prefetch(nxt + (size_t)tid * 96 + 64, 0, 0);
        }

        // A fragments: this WG's 16 h rows from the LDS mirror
        const __bf16* hrow = hcur + (size_t)n * Hh;
        v16bf a[6];
#pragma unroll
        for (int ks = 0; ks < 6; ++ks) a[ks] = load_frag(hrow + ks * 32, half);

        v8f accr, accz, accn;
#pragma unroll
        for (int v = 0; v < 8; ++v) { accr[v] = 0.f; accz[v] = 0.f; accn[v] = 0.f; }
#pragma unroll
        for (int ks = 0; ks < 6; ++ks) accr = wmma_bf16(a[ks], bfrag[0][ks], accr);
#pragma unroll
        for (int ks = 0; ks < 6; ++ks) accz = wmma_bf16(a[ks], bfrag[1][ks], accz);
#pragma unroll
        for (int ks = 0; ks < 6; ++ks) accn = wmma_bf16(a[ks], bfrag[2][ks], accn);

#pragma unroll
        for (int v = 0; v < 8; ++v) {
            const int rl = v + 8 * half;             // local batch row 0..15
            const float* gr = gxt + (size_t)rl * G3H + j0 + n;
            const float xr = gr[0];
            const float xz = gr[Hh];
            const float xn = gr[2 * Hh];
            const float rr = sigmoidf_(xr + accr[v] + bb[0]);
            const float zz = sigmoidf_(xz + accz[v] + bb[1]);
            const float nn = tanhf(xn + rr * (accn[v] + bb[2]));
            hreg[v] = (1.0f - zz) * nn + zz * hreg[v];

            const __bf16 hb = (__bf16)hreg[v];
            hnxt[(size_t)rl * Hh + j0 + n] = hb;     // safe: hnxt fully consumed
                                                     // before barrier(s-1)
            out[(size_t)t * Bb * (2 * Hh) + (size_t)(b0 + rl) * (2 * Hh) +
                dir * Hh + j0 + n] = hb;
        }
        __syncthreads();   // single barrier per step (ping-pong buffers)
    }
}

// ---------------------------------------------------------------------------
// Head: y[b] = fc2( relu( fc1( h2[T-1, b, :] ) ) )
// ---------------------------------------------------------------------------
__global__ void head_kernel(const __bf16* __restrict__ h2, const float* __restrict__ fc1w,
                            const float* __restrict__ fc1b, const float* __restrict__ fc2w,
                            const float* __restrict__ fc2b, float* __restrict__ out) {
    const int b = threadIdx.x;   // 0..127
    const __bf16* last = h2 + (size_t)(Tt - 1) * Bb * (2 * Hh) + (size_t)b * (2 * Hh);
    float acc = fc2b[0];
    for (int j = 0; j < 128; ++j) {
        float s = fc1b[j];
        const float* w = fc1w + (size_t)j * (2 * Hh);
        for (int k = 0; k < 2 * Hh; ++k) s += w[k] * (float)last[k];
        s = s > 0.0f ? s : 0.0f;
        acc += fc2w[j] * s;
    }
    out[b] = acc;
}

// ---------------------------------------------------------------------------
extern "C" void kernel_launch(void* const* d_in, const int* in_sizes, int n_in,
                              void* d_out, int out_size, void* d_ws, size_t ws_size,
                              hipStream_t stream) {
    (void)in_sizes; (void)n_in; (void)out_size; (void)ws_size;

    const int*   x    = (const int*)d_in[0];
    const float* emb  = (const float*)d_in[1];
    const float* w_ih[3] = {(const float*)d_in[2], (const float*)d_in[6],  (const float*)d_in[10]};
    const float* w_hh[3] = {(const float*)d_in[3], (const float*)d_in[7],  (const float*)d_in[11]};
    const float* b_ih[3] = {(const float*)d_in[4], (const float*)d_in[8],  (const float*)d_in[12]};
    const float* b_hh[3] = {(const float*)d_in[5], (const float*)d_in[9],  (const float*)d_in[13]};
    const float* fc1w = (const float*)d_in[14];
    const float* fc1b = (const float*)d_in[15];
    const float* fc2w = (const float*)d_in[16];
    const float* fc2b = (const float*)d_in[17];
    float* out = (float*)d_out;

    const size_t M = (size_t)Tt * Bb;            // 65536

    // workspace carve-up (256-B aligned)
    char* ws = (char*)d_ws;
    size_t off = 0;
    auto carve = [&](size_t bytes) -> char* {
        char* p = ws + off;
        off += (bytes + 255) & ~(size_t)255;
        return p;
    };
    __bf16* act0   = (__bf16*)carve(M * Ee * sizeof(__bf16));            // 16.8 MB
    __bf16* io0    = (__bf16*)carve(M * 2 * Hh * sizeof(__bf16));        // 50.3 MB
    __bf16* io1    = (__bf16*)carve(M * 2 * Hh * sizeof(__bf16));        // 50.3 MB
    float*  gx     = (float*) carve(2 * M * G3H * sizeof(float));        // 302 MB
    __bf16* wih_bf = (__bf16*)carve((size_t)2 * G3H * 384 * sizeof(__bf16));
    __bf16* whh_bf = (__bf16*)carve((size_t)2 * G3H * Hh * sizeof(__bf16));

    // 1) embedding gather -> bf16 (T,B,E)
    {
        const size_t tot = M * Ee;
        embed_kernel<<<dim3((unsigned)((tot + 255) / 256)), dim3(256), 0, stream>>>(
            x, emb, act0);
    }

    const int din[3]      = {Ee, 2 * Hh, 2 * Hh};
    const __bf16* lin[3]  = {act0, io0, io1};
    __bf16*       lout[3] = {io0, io1, io0};     // layer2 reuses io0

    for (int l = 0; l < 3; ++l) {
        const int K = din[l];
        // convert this layer's weights to bf16
        {
            int nih = 2 * G3H * K;
            to_bf16_kernel<<<dim3((nih + 255) / 256), dim3(256), 0, stream>>>(
                w_ih[l], wih_bf, nih);
            int nhh = 2 * G3H * Hh;
            to_bf16_kernel<<<dim3((nhh + 255) / 256), dim3(256), 0, stream>>>(
                w_hh[l], whh_bf, nhh);
        }
        // gx[dir] = act @ w_ih[dir]^T + b_ih[dir]   (both dirs via grid.y)
        {
            const int ngroups = G3H / 64;                      // 9
            const unsigned blocks = (unsigned)((M / 16) * ngroups / 8);  // 4608
            gemm_proj_kernel<<<dim3(blocks, 2), dim3(256), 0, stream>>>(
                lin[l], wih_bf, b_ih[l], gx, (int)M, G3H, K);
        }
        // persistent batch-split bidirectional scan (16 WGs)
        gru_scan_kernel<<<dim3(8, 2), dim3(384), 0, stream>>>(gx, whh_bf, b_hh[l],
                                                              lout[l]);
    }

    head_kernel<<<dim3(1), dim3(128), 0, stream>>>(lout[2], fc1w, fc1b, fc2w, fc2b, out);
}
